// CrossAttention_57964878627478
// MI455X (gfx1250) — matrix operand order repaired and checked by reading, not compiled, not run on hardware
//
#include <hip/hip_runtime.h>
#include <hip/hip_bf16.h>

// ---------- types ----------
typedef __attribute__((ext_vector_type(16))) __bf16       v16bf;
typedef __attribute__((ext_vector_type(8)))  float        v8f;
typedef __attribute__((ext_vector_type(4)))  unsigned int u32x4;
typedef __attribute__((ext_vector_type(8)))  unsigned int u32x8;

#define HW   4096
#define DIM  256
#define DQK  32
#define NK   64          // keys per tile
#define NT   (HW / NK)   // 64 key tiles
#define KPK  40          // K LDS row pitch (ushorts): 32 ch + 8 pad (80B rows)
#define VPP  72          // V LDS row pitch (ushorts): 64 keys + 8 pad (144B rows)

static __device__ __forceinline__ unsigned short f32_to_bf16(float f) {
  unsigned int u = __builtin_bit_cast(unsigned int, f);
  u += 0x7FFFu + ((u >> 16) & 1u);          // round-to-nearest-even
  return (unsigned short)(u >> 16);
}

static __device__ __forceinline__ unsigned int pack_bf16(float lo, float hi) {
#if __has_builtin(__builtin_amdgcn_cvt_pk_bf16_f32)
  auto t = __builtin_amdgcn_cvt_pk_bf16_f32(lo, hi);
  return __builtin_bit_cast(unsigned int, t);
#else
  return (unsigned int)f32_to_bf16(lo) | ((unsigned int)f32_to_bf16(hi) << 16);
#endif
}

static __device__ __forceinline__ v16bf make_v16bf(u32x4 a, u32x4 b) {
  u32x8 c;
  c[0] = a[0]; c[1] = a[1]; c[2] = a[2]; c[3] = a[3];
  c[4] = b[0]; c[5] = b[1]; c[6] = b[2]; c[7] = b[3];
  return __builtin_bit_cast(v16bf, c);
}

// ---------------------------------------------------------------------------
// Projection: q/k/v = W @ X + b for both sources.  Outputs bf16:
//   QT, KT : [src][b][HW][32]   (token-major -> contiguous WMMA operand rows)
//   Vb     : [src][b][256][HW]  (channel-major -> contiguous key runs)
// ---------------------------------------------------------------------------
__global__ __launch_bounds__(256) void proj_kernel(
    const float* __restrict__ x,  const float* __restrict__ y,
    const float* __restrict__ Wq, const float* __restrict__ bq,
    const float* __restrict__ Wk, const float* __restrict__ bk,
    const float* __restrict__ Wv, const float* __restrict__ bv,
    unsigned short* __restrict__ QT, unsigned short* __restrict__ KT,
    unsigned short* __restrict__ Vb) {
  __shared__ float Wrow[DIM];
  const int tid = threadIdx.x;
  const unsigned bid = blockIdx.x;           // ((s*4+b)*320 + o)*16 + ib
  const int ib = bid & 15;
  const int o  = (bid >> 4) % 320;
  const int sb = (bid >> 4) / 320;           // s*4 + b
  const int b  = sb & 3;
  const int s  = sb >> 2;

  const float* Wsel; float bias;
  if (o < 32)      { Wsel = Wq + (size_t)o * DIM;        bias = bq[o];      }
  else if (o < 64) { Wsel = Wk + (size_t)(o - 32) * DIM; bias = bk[o - 32]; }
  else             { Wsel = Wv + (size_t)(o - 64) * DIM; bias = bv[o - 64]; }
  Wrow[tid] = Wsel[tid];
  __syncthreads();

  const int i = ib * 256 + tid;
  const float* X = (s == 0 ? x : y) + (size_t)b * DIM * HW + i;
  float acc = bias;
  #pragma unroll 4
  for (int c = 0; c < DIM; ++c) acc += Wrow[c] * X[(size_t)c * HW];

  const unsigned short hv = f32_to_bf16(acc);
  if (o < 32)      QT[((size_t)sb * HW + i) * DQK + o]          = hv;
  else if (o < 64) KT[((size_t)sb * HW + i) * DQK + (o - 32)]   = hv;
  else             Vb[((size_t)sb * DIM + (o - 64)) * HW + i]   = hv;
}

// ---------------------------------------------------------------------------
// Flash cross-attention.  Grid: dir(2) x b(4) x 64-query blocks (64) = 512 WGs.
// WG = 8 waves: wave = (qtile[0..3], chan-half[0..1]); each wave owns
// 16 queries x 128 output channels.  Sᵀ trick keeps softmax in-lane.
// K/V tiles (64 keys) stream via async global->LDS, double buffered.
// ---------------------------------------------------------------------------
__global__ __launch_bounds__(256) void attn_kernel(
    const unsigned short* __restrict__ QT,
    const unsigned short* __restrict__ KT,
    const unsigned short* __restrict__ Vb,
    float* __restrict__ out) {
  __shared__ __align__(16) unsigned short Klds[2][NK * KPK];   // 2 x 64keys x 32ch
  __shared__ __align__(16) unsigned short Vlds[2][DIM * VPP];  // 2 x 256ch x 64keys

  const int tid  = threadIdx.x;
  const int lane = tid & 31;
  const int wave = tid >> 5;
  const int lm   = lane & 15;    // query index / matrix N-lane
  const int lh   = lane >> 4;    // lane half

  const unsigned bid = blockIdx.x;
  const int qb  = bid & 63;
  const int b   = (bid >> 6) & 3;
  const int dir = bid >> 8;      // 0: q from x, k/v from y ; 1: swapped
  const int sq  = dir;
  const int skv = 1 - dir;

  const unsigned short* Qbase = QT + (size_t)(sq  * 4 + b) * HW * DQK;
  const unsigned short* Kbase = KT + (size_t)(skv * 4 + b) * HW * DQK;
  const unsigned short* Vbase = Vb + (size_t)(skv * 4 + b) * DIM * HW;
  float*                Obase = out + (size_t)(dir * 4 + b) * DIM * HW;

  const int qt    = wave >> 1;
  const int chalf = wave & 1;
  const int q0    = qb * 64 + qt * 16;
  const int c0w   = chalf * 128;

  // -- async tile loader: exactly 9 x b128 per thread (uniform per wave) --
  const int kr_ = tid >> 2, kc_ = tid & 3;           // K: 64 rows x 4 chunks
  auto issue_tile = [&](int buf, int j0) {
    {
      unsigned ldsa = (unsigned)(size_t)&Klds[buf][kr_ * KPK + kc_ * 8];
      unsigned voff = (unsigned)(((j0 + kr_) * DQK + kc_ * 8) * 2);
      asm volatile("global_load_async_to_lds_b128 %0, %1, %2"
                   :: "v"(ldsa), "v"(voff), "s"(Kbase) : "memory");
    }
    #pragma unroll
    for (int rep = 0; rep < 8; ++rep) {              // V: 256 rows x 8 chunks
      const int idx = rep * 256 + tid;
      const int c = idx >> 3, ch = idx & 7;
      unsigned ldsa = (unsigned)(size_t)&Vlds[buf][c * VPP + ch * 8];
      unsigned voff = (unsigned)((c * HW + j0 + ch * 8) * 2);
      asm volatile("global_load_async_to_lds_b128 %0, %1, %2"
                   :: "v"(ldsa), "v"(voff), "s"(Vbase) : "memory");
    }
  };

  // Q as WMMA B-operand (32ch x 16 queries), resident for the whole kernel.
  const unsigned short* qrow = Qbase + (size_t)(q0 + lm) * DQK + lh * 16;
  const v16bf Bq = make_v16bf(*(const u32x4*)qrow, *(const u32x4*)(qrow + 8));

  v8f acc[8];
  #pragma unroll
  for (int t = 0; t < 8; ++t)
    #pragma unroll
    for (int r = 0; r < 8; ++r) acc[t][r] = 0.0f;
  float m_run = -3.0e38f, l_run = 0.0f;

  issue_tile(0, 0);
  issue_tile(1, NK);

  for (int kt = 0; kt < NT; ++kt) {
    // previous batch (9 async ops of the NEXT tile) may stay in flight
    if (kt < NT - 1) asm volatile("s_wait_asynccnt 0x9" ::: "memory");
    else             asm volatile("s_wait_asynccnt 0x0" ::: "memory");
    __syncthreads();

    const unsigned short* Kb = &Klds[kt & 1][0];
    const unsigned short* VB = &Vlds[kt & 1][0];

    // ---- S^T[j][i] = sum_c K[c][j]·Q[c][i]  (4 x 16-key subtiles, K-dim=32)
    v8f z;
    #pragma unroll
    for (int r = 0; r < 8; ++r) z[r] = 0.0f;
    v8f st[4];
    #pragma unroll
    for (int js = 0; js < 4; ++js) {
      const unsigned short* kr = &Kb[(js * 16 + lm) * KPK];
      const v16bf Ak = make_v16bf(*(const u32x4*)(kr + lh * 16),
                                  *(const u32x4*)(kr + lh * 16 + 8));
      st[js] = __builtin_amdgcn_wmma_f32_16x16x32_bf16(
          false, Ak, false, Bq, (short)0, z, false, false);
    }

    // ---- online softmax over 64 keys (query i = lane%16; j split across
    //      the two lane halves -> one shfl_xor(16))
    float lmax = st[0][0];
    #pragma unroll
    for (int js = 0; js < 4; ++js)
      #pragma unroll
      for (int r = 0; r < 8; ++r) lmax = fmaxf(lmax, st[js][r]);
    lmax = fmaxf(lmax, __shfl_xor(lmax, 16, 32));
    const float mnew = fmaxf(m_run, lmax);

    float p[4][8], lsum = 0.0f;
    #pragma unroll
    for (int js = 0; js < 4; ++js)
      #pragma unroll
      for (int r = 0; r < 8; ++r) {
        p[js][r] = __expf(st[js][r] - mnew);
        lsum += p[js][r];
      }
    lsum += __shfl_xor(lsum, 16, 32);

    const float alpha = __expf(m_run - mnew);
    l_run = l_run * alpha + lsum;
    m_run = mnew;
    #pragma unroll
    for (int t = 0; t < 8; ++t)
      #pragma unroll
      for (int r = 0; r < 8; ++r) acc[t][r] *= alpha;

    // ---- repack P^T into two WMMA B-operands (keys 0..31 and 32..63)
    v16bf Pb[2];
    #pragma unroll
    for (int hb = 0; hb < 2; ++hb) {       // hb=0: st[0],st[1]; hb=1: st[2],st[3]
      unsigned int pk0[4], pk1[4];
      #pragma unroll
      for (int v = 0; v < 4; ++v) {
        pk0[v] = pack_bf16(p[2 * hb + 0][2 * v], p[2 * hb + 0][2 * v + 1]);
        pk1[v] = pack_bf16(p[2 * hb + 1][2 * v], p[2 * hb + 1][2 * v + 1]);
      }
      u32x8 buv;
      #pragma unroll
      for (int v = 0; v < 4; ++v) {
        const unsigned int x0 = __shfl_xor(pk0[v], 16, 32);
        const unsigned int x1 = __shfl_xor(pk1[v], 16, 32);
        buv[v]     = (lh == 0) ? pk0[v] : x1;
        buv[4 + v] = (lh == 0) ? x0     : pk1[v];
      }
      Pb[hb] = __builtin_bit_cast(v16bf, buv);
    }

    // ---- acc^T[c][i] += sum_j V[c][j]·P^T[j][i]  (8 ch-tiles x 2 k-chunks)
    #pragma unroll
    for (int t = 0; t < 8; ++t) {
      const unsigned short* vrow = &VB[(c0w + t * 16 + lm) * VPP];
      const v16bf Av0 = make_v16bf(*(const u32x4*)(vrow + lh * 16),
                                   *(const u32x4*)(vrow + lh * 16 + 8));
      acc[t] = __builtin_amdgcn_wmma_f32_16x16x32_bf16(
          false, Av0, false, Pb[0], (short)0, acc[t], false, false);
      const v16bf Av1 = make_v16bf(*(const u32x4*)(vrow + 32 + lh * 16),
                                   *(const u32x4*)(vrow + 40 + lh * 16));
      acc[t] = __builtin_amdgcn_wmma_f32_16x16x32_bf16(
          false, Av1, false, Pb[1], (short)0, acc[t], false, false);
    }
    __syncthreads();                       // all waves done with buf kt&1
    if (kt + 2 < NT) issue_tile(kt & 1, (kt + 2) * NK);
  }

  // ---- normalize and store: D layout row r -> channel r + 8*lh, N -> query
  const float inv = 1.0f / l_run;
  #pragma unroll
  for (int t = 0; t < 8; ++t) {
    #pragma unroll
    for (int r = 0; r < 8; ++r) {
      const int c = c0w + t * 16 + r + lh * 8;
      Obase[(size_t)c * HW + q0 + lm] = acc[t][r] * inv;
    }
  }
}

// ---------------------------------------------------------------------------
extern "C" void kernel_launch(void* const* d_in, const int* in_sizes, int n_in,
                              void* d_out, int out_size, void* d_ws, size_t ws_size,
                              hipStream_t stream) {
  const float* x  = (const float*)d_in[0];
  const float* y  = (const float*)d_in[1];
  const float* Wq = (const float*)d_in[2];
  const float* bq = (const float*)d_in[3];
  const float* Wk = (const float*)d_in[4];
  const float* bk = (const float*)d_in[5];
  const float* Wv = (const float*)d_in[6];
  const float* bv = (const float*)d_in[7];

  // workspace layout (bf16): QT | KT | V   (~20 MB total)
  unsigned short* ws = (unsigned short*)d_ws;
  unsigned short* QT = ws;                              // 2*4*4096*32
  unsigned short* KT = ws + (size_t)2 * 4 * HW * DQK;
  unsigned short* Vw = KT + (size_t)2 * 4 * HW * DQK;   // 2*4*256*4096

  // projections: 2 srcs * 4 batch * 320 rows * 16 pixel-blocks of 256
  proj_kernel<<<2 * 4 * 320 * 16, 256, 0, stream>>>(
      x, y, Wq, bq, Wk, bk, Wv, bv, QT, KT, Vw);

  // attention: 2 dirs * 4 batch * 64 query-blocks
  attn_kernel<<<2 * 4 * 64, 256, 0, stream>>>(QT, KT, Vw, (float*)d_out);
}
